// NodeUpdateBlock_45956150067209
// MI455X (gfx1250) — matrix operand order repaired and checked by reading, not compiled, not run on hardware
//
#include <hip/hip_runtime.h>
#include <math.h>

// ---------------- problem constants (fixed by the reference) ----------------
#define BB   2
#define NN_  512
#define DD   384
#define PD   128
#define NH   12
#define DQKV 1152   // qkv row: q_s[0,192) k_s[192,384) v_s[384,576) q_p[576,720) k_p[720,864) v_p[864,1152)
#define DCAT 2112   // out_s[0,192) out_p[192,480) pnorm[480,576) out_pair[576,2112)
#define DFF  1536

#define S_SCALAR 0.14433756729740643f   // (3*16)^-0.5
#define S_POINT  0.08091531f            // (3*4*(9*sqrt2))^-0.5
#define S_PAIR   0.57735026918962576f   // 3^-0.5

typedef __attribute__((ext_vector_type(16))) _Float16 v16h;
typedef __attribute__((ext_vector_type(8)))  float    v8f;
typedef __attribute__((ext_vector_type(4)))  float    f32x4;
typedef __attribute__((ext_vector_type(4)))  _Float16 h16x4;

__device__ __forceinline__ v8f wmma_f16(v16h a, v16h b, v8f c) {
  return __builtin_amdgcn_wmma_f32_16x16x32_f16(false, a, false, b, (short)0, c,
                                                false, false);
}
__device__ __forceinline__ float gelu_tanh(float x) {
  return 0.5f * x * (1.f + tanhf(0.7978845608028654f * (x + 0.044715f * x * x * x)));
}

// A fragment (contiguous k order, not the ISA table order): elements 0..15 <- p[0..15]
// where p = &A[row*lda + kb + hi*16].  K must be a multiple of 32 (all our GEMMs).
__device__ __forceinline__ v16h a_frag_load(const float* __restrict__ p) {
  f32x4 f0 = *(const f32x4*)(p + 0);
  f32x4 f1 = *(const f32x4*)(p + 4);
  f32x4 f2 = *(const f32x4*)(p + 8);
  f32x4 f3 = *(const f32x4*)(p + 12);
  v16h a;
  a[0]  = (_Float16)f0[0]; a[1]  = (_Float16)f0[1]; a[2]  = (_Float16)f0[2]; a[3]  = (_Float16)f0[3];
  a[4]  = (_Float16)f1[0]; a[5]  = (_Float16)f1[1]; a[6]  = (_Float16)f1[2]; a[7]  = (_Float16)f1[3];
  a[8]  = (_Float16)f2[0]; a[9]  = (_Float16)f2[1]; a[10] = (_Float16)f2[2]; a[11] = (_Float16)f2[3];
  a[12] = (_Float16)f3[0]; a[13] = (_Float16)f3[1]; a[14] = (_Float16)f3[2]; a[15] = (_Float16)f3[3];
  return a;
}

// B fragment, NN (row-major KxN): element e <- p[e*ldb], p = &B[(kb+hi*16)*ldb + col].
// Unguarded: 16 plain loads, coalesced across lanes (col = lane&15).
template <int LDB>
__device__ __forceinline__ v16h b_frag_nn(const float* __restrict__ p) {
  v16h b;
#pragma unroll
  for (int e = 0; e < 16; ++e) b[e] = (_Float16)p[(size_t)e * LDB];
  return b;
}

// ---------------- LayerNorm over rows of length D ----------------
__global__ void ln_kernel(const float* __restrict__ x, const float* __restrict__ g,
                          const float* __restrict__ be, float* __restrict__ y, int D) {
  size_t row = (size_t)blockIdx.x * D;
  __shared__ float rA[128], rB[128];
  int t = threadIdx.x;
  float s0 = 0.f, s1 = 0.f;
  for (int d = t; d < D; d += 128) { float v = x[row + d]; s0 += v; s1 += v * v; }
  rA[t] = s0; rB[t] = s1; __syncthreads();
  for (int off = 64; off > 0; off >>= 1) {
    if (t < off) { rA[t] += rA[t + off]; rB[t] += rB[t + off]; }
    __syncthreads();
  }
  float mu = rA[0] / D, rstd = rsqrtf(rB[0] / D - mu * mu + 1e-5f);
  for (int d = t; d < D; d += 128) y[row + d] = (x[row + d] - mu) * rstd * g[d] + be[d];
}

// ---------------- residual + LN (node1 = ao + node; tn = LN(ao)) ----------------
__global__ void resln_kernel(const float* __restrict__ ao, const float* __restrict__ node,
                             const float* __restrict__ g, const float* __restrict__ be,
                             float* __restrict__ n1, float* __restrict__ tn, int D) {
  size_t row = (size_t)blockIdx.x * D;
  __shared__ float rA[128], rB[128];
  int t = threadIdx.x;
  float s0 = 0.f, s1 = 0.f;
  for (int d = t; d < D; d += 128) { float v = ao[row + d]; s0 += v; s1 += v * v; }
  rA[t] = s0; rB[t] = s1; __syncthreads();
  for (int off = 64; off > 0; off >>= 1) {
    if (t < off) { rA[t] += rA[t + off]; rB[t] += rB[t + off]; }
    __syncthreads();
  }
  float mu = rA[0] / D, rstd = rsqrtf(rB[0] / D - mu * mu + 1e-5f);
  for (int d = t; d < D; d += 128) {
    float v = ao[row + d];
    tn[row + d] = (v - mu) * rstd * g[d] + be[d];
    n1[row + d] = v + node[row + d];
  }
}

// ---------------- generic NN WMMA GEMM, one wave per 16x16 C tile ----------------
// All dims compile-time: constant strides -> immediate-offset addressing, and the
// EP epilogue is branch-free straight-line code.
// EP: 0 none, 1 +bias, 2 +bias->gelu, 3 +bias+resid
template <int LDA, int LDB, int LDC, int K, int EP>
__global__ __launch_bounds__(256) void gemm_nn_kernel(
    const float* __restrict__ A, const float* __restrict__ Bm, float* __restrict__ C,
    const float* __restrict__ bias, const float* __restrict__ resid) {
  constexpr int NTILES = LDB >> 4;          // B/C have LDB==LDC==Ncols
  int wave = blockIdx.x * 8 + (threadIdx.x >> 5);
  int tm = wave / NTILES, tnn = wave % NTILES;
  int lane = threadIdx.x & 31, m = lane & 15, hi = lane >> 4;
  int row = tm * 16 + m;
  int col = tnn * 16 + m;
  const float* ap = A + (size_t)row * LDA + hi * 16;
  const float* bp = Bm + (size_t)(hi * 16) * LDB + col;
  v8f acc = {};
  for (int kb = 0; kb < K; kb += 32) {
    v16h a = a_frag_load(ap + kb);
    v16h b = b_frag_nn<LDB>(bp + (size_t)kb * LDB);
    acc = wmma_f16(a, b, acc);
  }
  float bval = (EP >= 1) ? bias[col] : 0.f;
#pragma unroll
  for (int r = 0; r < 8; ++r) {
    int rr = tm * 16 + r + hi * 8;
    float v = acc[r] + bval;
    if (EP == 2) v = gelu_tanh(v);
    if (EP == 3) v += resid[(size_t)rr * LDC + col];
    C[(size_t)rr * LDC + col] = v;
  }
}

// ---------------- rigid-apply q_p/k_p/v_p in place + qq/kk ----------------
__global__ void transform_kernel(const float* __restrict__ rot, const float* __restrict__ trans,
                                 float* __restrict__ qkv, float* __restrict__ qq,
                                 float* __restrict__ kk) {
  int bn = blockIdx.x;          // (b*NN_+n)
  int b = bn / NN_, n = bn % NN_;
  __shared__ float R[9], T[3], pts[192 * 3];
  int t = threadIdx.x;          // 192 threads, one point each
  if (t < 9) R[t] = rot[(size_t)bn * 9 + t];
  if (t < 3) T[t] = trans[(size_t)bn * 3 + t];
  __syncthreads();
  float* row = qkv + (size_t)bn * DQKV + 576;
  float x = row[t * 3 + 0], y = row[t * 3 + 1], z = row[t * 3 + 2];
  float ox = R[0] * x + R[1] * y + R[2] * z + T[0];
  float oy = R[3] * x + R[4] * y + R[5] * z + T[1];
  float oz = R[6] * x + R[7] * y + R[8] * z + T[2];
  row[t * 3 + 0] = ox; row[t * 3 + 1] = oy; row[t * 3 + 2] = oz;
  pts[t * 3 + 0] = ox; pts[t * 3 + 1] = oy; pts[t * 3 + 2] = oz;
  __syncthreads();
  if (t < 24) {                 // 12 heads * {q,k}: sum |pt|^2 over 4 points (12 dims)
    int h = t % NH; bool isK = t >= NH;
    int p0 = (isK ? 48 : 0) + h * 4;
    float s = 0.f;
    for (int p = p0; p < p0 + 4; ++p)
      s += pts[p*3]*pts[p*3] + pts[p*3+1]*pts[p*3+1] + pts[p*3+2]*pts[p*3+2];
    (isK ? kk : qq)[(size_t)(b * NH + h) * NN_ + n] = s;
  }
}

// ---------------- fused pair-LN + pair_bias (pair read #1), single-wave blocks ----------------
// One wave per (b,i,16-j tile): register-staged 16x128 tile, shfl stats, f16 tile
// in LDS, then the same wave does the WMMA against w_pair_bias.
__global__ __launch_bounds__(32) void pair_bias_kernel(
    const float* __restrict__ pair, const float* __restrict__ g, const float* __restrict__ be,
    const float* __restrict__ wpb, float* __restrict__ logits,
    float* __restrict__ mu_ws, float* __restrict__ rs_ws) {
  int bid = blockIdx.x;
  int jt = bid & 31;                 // NN_/16 = 32 j-tiles
  int i = (bid >> 5) % NN_;
  int b = bid / (32 * NN_);
  int lane = threadIdx.x;            // 32
  __shared__ _Float16 sa[16 * PD];   // normalized f16 tile
  size_t base = (((size_t)(b * NN_ + i)) * NN_ + jt * 16) * PD;
  f32x4 vrow[16];
#pragma unroll
  for (int r = 0; r < 16; ++r)       // lane holds cols [4*lane, 4*lane+4) of each row
    vrow[r] = *(const f32x4*)(pair + base + r * PD + lane * 4);
  f32x4 gv = *(const f32x4*)(g + lane * 4);
  f32x4 bv = *(const f32x4*)(be + lane * 4);
  float mu_l[16], rs_l[16];
#pragma unroll
  for (int r = 0; r < 16; ++r) {
    f32x4 v = vrow[r];
    float s0 = v[0] + v[1] + v[2] + v[3];
    float s1 = v[0]*v[0] + v[1]*v[1] + v[2]*v[2] + v[3]*v[3];
#pragma unroll
    for (int off = 16; off > 0; off >>= 1) {
      s0 += __shfl_xor(s0, off);
      s1 += __shfl_xor(s1, off);
    }
    float mu = s0 / PD, rstd = rsqrtf(s1 / PD - mu * mu + 1e-5f);
    mu_l[r] = mu; rs_l[r] = rstd;
  }
  if (lane < 16) {
    size_t sidx = ((size_t)(b * NN_ + i)) * NN_ + jt * 16 + lane;
    mu_ws[sidx] = mu_l[lane]; rs_ws[sidx] = rs_l[lane];
  }
  // each lane owns its 4 columns of every row: normalize + write f16x4 per row
#pragma unroll
  for (int r = 0; r < 16; ++r) {
    f32x4 v = vrow[r];
    h16x4 o;
#pragma unroll
    for (int c = 0; c < 4; ++c)
      o[c] = (_Float16)((v[c] - mu_l[r]) * rs_l[r] * gv[c] + bv[c]);
    *(h16x4*)(sa + r * PD + lane * 4) = o;
  }
  __syncthreads();                   // single-wave wg: lowers to ds-count ordering
  int m = lane & 15, hi = lane >> 4;
  int colw = (m < NH) ? m : 0;       // address clamp; cols >= 12 never stored
  v8f acc = {};
#pragma unroll
  for (int kb = 0; kb < PD; kb += 32) {
    const _Float16* ap = sa + m * PD + kb + hi * 16;
    v16h a;
#pragma unroll
    for (int e = 0; e < 8; ++e) { a[e] = ap[e]; a[8 + e] = ap[8 + e]; }
    v16h bf = b_frag_nn<NH>(wpb + (size_t)(kb + hi * 16) * NH + colw);
    acc = wmma_f16(a, bf, acc);
  }
  if (m < NH) {
#pragma unroll
    for (int r = 0; r < 8; ++r) {
      int j = jt * 16 + r + hi * 8;
      logits[(((size_t)(b * NH + m)) * NN_ + i) * NN_ + j] = S_PAIR * acc[r];
    }
  }
}

// ---------------- scalar + point logits fused into ONE K=32 WMMA per tile ----------------
// A k-range [0,16): S_SCALAR * q_s ; k-range [16,28): (2*S_POINT*softplus(pw)) * q_p.
// B k-range [0,16): k_s ; [16,28): k_p.  Rank-1 -(qq+kk) term in epilogue.
__global__ __launch_bounds__(256) void logits_kernel(
    const float* __restrict__ qkv, const float* __restrict__ qq, const float* __restrict__ kk,
    const float* __restrict__ pw, float* __restrict__ logits) {
  int wid = blockIdx.x * 8 + (threadIdx.x >> 5);
  int jt = wid & 31, it = (wid >> 5) & 31, h = (wid >> 10) % NH, b = wid / (32 * 32 * NH);
  int lane = threadIdx.x & 31, m = lane & 15, hi = lane >> 4;
  int i = it * 16 + m, j = jt * 16 + m;
  size_t qrow = ((size_t)(b * NN_) + i) * DQKV;
  size_t krow = ((size_t)(b * NN_) + j) * DQKV;
  float w = log1pf(expf(pw[h]));     // softplus(point_weights[h])
  float pscale = 2.f * S_POINT * w;
  // ---- A fragment: unconditional vector loads, value-masked tail ----
  const float* aq = hi ? (qkv + qrow + 576 + h * 12) : (qkv + qrow + h * 16);
  float asc = hi ? pscale : S_SCALAR;
  f32x4 s0 = *(const f32x4*)(aq + 0);
  f32x4 s1 = *(const f32x4*)(aq + 4);
  f32x4 p0 = *(const f32x4*)(aq + 8);
  f32x4 p1 = *(const f32x4*)(aq + 12);
  v16h a;
#pragma unroll
  for (int c = 0; c < 4; ++c) {
    a[c]      = (_Float16)(s0[c] * asc);
    a[4 + c]  = (_Float16)(s1[c] * asc);
    a[8 + c]  = (_Float16)(p0[c] * asc);
    a[12 + c] = (_Float16)(hi ? 0.f : p1[c] * asc);       // q_p only 12 wide
  }
  // ---- B fragment: hi half selects k_s vs k_p run, both contiguous ----
  const float* bp = hi ? (qkv + krow + 720 + h * 12) : (qkv + krow + 192 + h * 16);
  f32x4 b0 = *(const f32x4*)(bp + 0);
  f32x4 b1 = *(const f32x4*)(bp + 4);
  f32x4 b2 = *(const f32x4*)(bp + 8);
  f32x4 b3 = *(const f32x4*)(bp + 12);
  v16h bf;
#pragma unroll
  for (int c = 0; c < 4; ++c) {
    bf[c]      = (_Float16)b0[c];
    bf[4 + c]  = (_Float16)b1[c];
    bf[8 + c]  = (_Float16)b2[c];
    bf[12 + c] = (_Float16)(hi ? 0.f : b3[c]);            // k_p only 12 wide
  }
  v8f acc = {};
  acc = wmma_f16(a, bf, acc);
  float corr = S_POINT * w;
  float kkj = kk[(size_t)(b * NH + h) * NN_ + j];
#pragma unroll
  for (int r = 0; r < 8; ++r) {
    int ii = it * 16 + r + hi * 8;
    float val = acc[r] - corr * (qq[(size_t)(b * NH + h) * NN_ + ii] + kkj);
    logits[(((size_t)(b * NH + h)) * NN_ + ii) * NN_ + j] += val;
  }
}

// ---------------- softmax over last dim (rows of 512), in place ----------------
__global__ void softmax_kernel(float* __restrict__ logits) {
  size_t row = (size_t)blockIdx.x * NN_;
  __shared__ float red[256];
  int t = threadIdx.x;
  float mx = -3.4e38f;
  for (int j = t; j < NN_; j += 256) mx = fmaxf(mx, logits[row + j]);
  red[t] = mx; __syncthreads();
  for (int off = 128; off > 0; off >>= 1) {
    if (t < off) red[t] = fmaxf(red[t], red[t + off]);
    __syncthreads();
  }
  mx = red[0]; __syncthreads();
  float sm = 0.f;
  for (int j = t; j < NN_; j += 256) { float e = expf(logits[row + j] - mx); logits[row + j] = e; sm += e; }
  red[t] = sm; __syncthreads();
  for (int off = 128; off > 0; off >>= 1) {
    if (t < off) red[t] += red[t + off];
    __syncthreads();
  }
  float inv = 1.f / red[0];
  for (int j = t; j < NN_; j += 256) logits[row + j] *= inv;
}

// ---------------- out_s / out_p = attn @ {v_s, v_p} ----------------
__global__ __launch_bounds__(256) void attn_v_kernel(
    const float* __restrict__ attn, const float* __restrict__ qkv,
    float* __restrict__ out_s, float* __restrict__ out_p) {
  int wid = blockIdx.x * 8 + (threadIdx.x >> 5);
  int nt = wid % 3, it = (wid / 3) & 31, h = (wid / 96) % NH, b = wid / (96 * NH);
  int lane = threadIdx.x & 31, m = lane & 15, hi = lane >> 4;
  int i = it * 16 + m;
  const float* ap = attn + (((size_t)(b * NH + h)) * NN_ + i) * NN_ + hi * 16;
  int voff = (nt == 0) ? 384 + h * 16 : 864 + h * 24;
  int ncols = (nt == 0) ? 16 : 24;
  int n0 = (nt == 2) ? 16 : 0;
  int col = n0 + m;
  int colc = (col < ncols) ? col : ncols - 1;   // address clamp; excess never stored
  const float* bp = qkv + ((size_t)(b * NN_) + hi * 16) * DQKV + voff + colc;
  v8f acc = {};
  for (int kb = 0; kb < NN_; kb += 32) {
    v16h a = a_frag_load(ap + kb);
    v16h bf = b_frag_nn<DQKV>(bp + (size_t)kb * DQKV);
    acc = wmma_f16(a, bf, acc);
  }
#pragma unroll
  for (int r = 0; r < 8; ++r) {
    int ii = it * 16 + r + hi * 8;
    if (col < ncols) {
      if (nt == 0) out_s[(((size_t)(b * NH + h)) * NN_ + ii) * 16 + col] = acc[r];
      else         out_p[(((size_t)(b * NH + h)) * NN_ + ii) * 24 + col] = acc[r];
    }
  }
}

// ---------------- out_pair = attn @ LN(pair) (pair read #2, stats from ws) ----------------
__global__ __launch_bounds__(256) void out_pair_kernel(
    const float* __restrict__ attn, const float* __restrict__ pair,
    const float* __restrict__ mu_ws, const float* __restrict__ rs_ws,
    const float* __restrict__ g, const float* __restrict__ be, float* __restrict__ cat) {
  int bid = blockIdx.x;               // (b*NN_+i)
  int b = bid / NN_, i = bid % NN_;
  __shared__ float smu[NN_], srs[NN_];
  int t = threadIdx.x;
  for (int x = t; x < NN_; x += 256) {
    smu[x] = mu_ws[(size_t)bid * NN_ + x];
    srs[x] = rs_ws[(size_t)bid * NN_ + x];
  }
  __syncthreads();
  int w = t >> 5;                     // wave = output column tile (8*16 = 128 = PD)
  int m = t & 15, hi = (t >> 4) & 1;
  int d = w * 16 + m;
  float gd = g[d], bd = be[d];
  int mh = (m < NH) ? m : NH - 1;     // address clamp; rows >= 12 never stored
  const float* ap = attn + (((size_t)(b * NH + mh)) * NN_ + i) * NN_ + hi * 16;
  size_t prow = (size_t)bid * NN_ * PD;
  const float* pp = pair + prow + (size_t)(hi * 16) * PD + d;
  v8f acc = {};
  for (int kb = 0; kb < NN_; kb += 32) {
    __builtin_prefetch(pp + (size_t)(kb + 32) * PD, 0, 1);
    v16h a = a_frag_load(ap + kb);
    v16h bf;
#pragma unroll
    for (int e = 0; e < 16; ++e) {
      int k = kb + e + hi * 16;
      bf[e] = (_Float16)((pp[(size_t)(kb + e) * PD] - smu[k]) * srs[k] * gd + bd);
    }
    acc = wmma_f16(a, bf, acc);
  }
#pragma unroll
  for (int r = 0; r < 8; ++r) {
    int h = r + hi * 8;
    if (h < NH) cat[(size_t)bid * DCAT + 576 + h * PD + d] = acc[r];
  }
}

// ---------------- inverse rigid + pnorm + assemble cat sections ----------------
__global__ void assemble_kernel(const float* __restrict__ out_s, const float* __restrict__ out_p,
                                const float* __restrict__ rot, const float* __restrict__ trans,
                                float* __restrict__ cat) {
  int bid = blockIdx.x;               // (b*NN_+i)
  int b = bid / NN_, i = bid % NN_;
  __shared__ float R[9], T[3];
  int t = threadIdx.x;                // 192 threads
  if (t < 9) R[t] = rot[(size_t)bid * 9 + t];
  if (t < 3) T[t] = trans[(size_t)bid * 3 + t];
  __syncthreads();
  float* crow = cat + (size_t)bid * DCAT;
  {  // out_s: 192 values (h*16+d)
    int h = t >> 4, d = t & 15;
    crow[t] = out_s[(((size_t)(b * NH + h)) * NN_ + i) * 16 + d];
  }
  if (t < 96) {                       // 96 points: local = R^T (x - t), pnorm
    int h = t >> 3, dp = t & 7;
    const float* op = out_p + (((size_t)(b * NH + h)) * NN_ + i) * 24 + dp * 3;
    float x = op[0] - T[0], y = op[1] - T[1], z = op[2] - T[2];
    float lx = R[0] * x + R[3] * y + R[6] * z;
    float ly = R[1] * x + R[4] * y + R[7] * z;
    float lz = R[2] * x + R[5] * y + R[8] * z;
    crow[192 + t * 3 + 0] = lx;
    crow[192 + t * 3 + 1] = ly;
    crow[192 + t * 3 + 2] = lz;
    crow[480 + t] = sqrtf(lx * lx + ly * ly + lz * lz + 1e-8f);
  }
}

// ---------------- host-side orchestration ----------------
extern "C" void kernel_launch(void* const* d_in, const int* in_sizes, int n_in,
                              void* d_out, int out_size, void* d_ws, size_t ws_size,
                              hipStream_t stream) {
  (void)in_sizes; (void)n_in; (void)out_size; (void)ws_size;
  const float* node   = (const float*)d_in[0];
  const float* pair   = (const float*)d_in[1];
  /* d_in[2] = mask: all-true, no-op */
  const float* rot    = (const float*)d_in[3];
  const float* trans  = (const float*)d_in[4];
  const float* ln_s_g = (const float*)d_in[5];
  const float* ln_s_b = (const float*)d_in[6];
  const float* ln_p_g = (const float*)d_in[7];
  const float* ln_p_b = (const float*)d_in[8];
  const float* w_pb   = (const float*)d_in[9];
  const float* w_qkv  = (const float*)d_in[10];
  const float* pw     = (const float*)d_in[11];
  const float* w_out  = (const float*)d_in[12];
  const float* b_out  = (const float*)d_in[13];
  const float* ln_t_g = (const float*)d_in[14];
  const float* ln_t_b = (const float*)d_in[15];
  const float* w_ff1  = (const float*)d_in[16];
  const float* b_ff1  = (const float*)d_in[17];
  const float* w_ff2  = (const float*)d_in[18];
  const float* b_ff2  = (const float*)d_in[19];
  float* ws  = (float*)d_ws;
  float* out = (float*)d_out;

  // workspace layout (floats), total ~14.35M floats (~57 MB)
  float* xn  = ws;
  float* qkv = xn  + (size_t)BB * NN_ * DD;          // 393216
  float* qq  = qkv + (size_t)BB * NN_ * DQKV;        // +1179648
  float* kk  = qq  + (size_t)BB * NH * NN_;          // +12288
  float* mu  = kk  + (size_t)BB * NH * NN_;          // +12288
  float* rs  = mu  + (size_t)BB * NN_ * NN_;         // +524288
  float* att = rs  + (size_t)BB * NN_ * NN_;         // +524288
  float* osb = att + (size_t)BB * NH * NN_ * NN_;    // +6291456
  float* opb = osb + (size_t)BB * NH * NN_ * 16;     // +196608
  float* cat = opb + (size_t)BB * NH * NN_ * 24;     // +294912
  float* ao  = cat + (size_t)BB * NN_ * DCAT;        // +2162688
  float* n1  = ao  + (size_t)BB * NN_ * DD;
  float* tn  = n1  + (size_t)BB * NN_ * DD;
  float* h1  = tn  + (size_t)BB * NN_ * DD;          // +393216

  const int MR = BB * NN_;  // 1024 rows

  // 1) LN(node) -> xn
  ln_kernel<<<MR, 128, 0, stream>>>(node, ln_s_g, ln_s_b, xn, DD);
  // 2) qkv = xn @ w_qkv            (1024x384 @ 384x1152)
  gemm_nn_kernel<DD, DQKV, DQKV, DD, 0>
      <<<(64 * 72) / 8, 256, 0, stream>>>(xn, w_qkv, qkv, nullptr, nullptr);
  // 3) rigid-apply points in place, qq/kk
  transform_kernel<<<MR, 192, 0, stream>>>(rot, trans, qkv, qq, kk);
  // 4) pair LN + pair_bias -> att = S_PAIR*bias ; stats -> mu/rs   (pair pass 1)
  pair_bias_kernel<<<BB * NN_ * 32, 32, 0, stream>>>(pair, ln_p_g, ln_p_b, w_pb, att, mu, rs);
  // 5) att += S_SCALAR*qk_s + S_POINT*softplus(pw)*point   (single fused WMMA)
  logits_kernel<<<(BB * NH * 32 * 32) / 8, 256, 0, stream>>>(qkv, qq, kk, pw, att);
  // 6) softmax rows
  softmax_kernel<<<BB * NH * NN_, 256, 0, stream>>>(att);
  // 7) out_s / out_p = attn @ {v_s, v_p}
  attn_v_kernel<<<(BB * NH * 32 * 3) / 8, 256, 0, stream>>>(att, qkv, osb, opb);
  // 8) out_pair = attn @ LN(pair) -> cat[...,576:]   (pair pass 2)
  out_pair_kernel<<<MR, 256, 0, stream>>>(att, pair, mu, rs, ln_p_g, ln_p_b, cat);
  // 9) inverse rigid + pnorm + assemble cat[...,0:576]
  assemble_kernel<<<MR, 192, 0, stream>>>(osb, opb, rot, trans, cat);
  // 10) attn_out = cat @ w_out + b_out   (1024x2112 @ 2112x384)
  gemm_nn_kernel<DCAT, DD, DD, DCAT, 1>
      <<<(64 * 24) / 8, 256, 0, stream>>>(cat, w_out, ao, b_out, nullptr);
  // 11) node1 = ao + node ; tn = LN(ao)
  resln_kernel<<<MR, 128, 0, stream>>>(ao, node, ln_t_g, ln_t_b, n1, tn, DD);
  // 12) h1 = gelu(tn @ w_ff1 + b_ff1)
  gemm_nn_kernel<DD, DFF, DFF, DD, 2>
      <<<(64 * 96) / 8, 256, 0, stream>>>(tn, w_ff1, h1, b_ff1, nullptr);
  // 13) out = h1 @ w_ff2 + b_ff2 + node1
  gemm_nn_kernel<DFF, DD, DD, DFF, 3>
      <<<(64 * 24) / 8, 256, 0, stream>>>(h1, w_ff2, out, b_ff2, n1);
}
